// HyperTransposeConv_88450556494405
// MI455X (gfx1250) — matrix operand order repaired and checked by reading, not compiled, not run on hardware
//
#include <hip/hip_runtime.h>
#include <hip/hip_bf16.h>

// ---------------------------------------------------------------------------
// HyperTransposeConv (Cayley-Dickson n=8) as 4 parity-class WMMA GEMMs.
//   x [8,512,64,64] f32, W [8,64,32,4,4] f32, b [8,32] f32
//   out [8,256,128,128] f32
// Per parity p=py*2+px:  C[co,pix] = sum_k A[co,k]*B[k,pix], K = 4 taps * 512.
// A tile staged by the Tensor Data Mover (async, TENSORcnt), B tile staged
// from pre-converted bf16 x; double-buffered LDS, bf16 WMMA w/ f32 accum.
// ---------------------------------------------------------------------------

typedef __attribute__((ext_vector_type(16))) __bf16 v16bf;
typedef __attribute__((ext_vector_type(4)))  __bf16 v4bf;
typedef __attribute__((ext_vector_type(8)))  float  v8f;
typedef __attribute__((ext_vector_type(4)))  unsigned int u32x4;
typedef __attribute__((ext_vector_type(8)))  int i32x8;
typedef __attribute__((ext_vector_type(4)))  int i32x4;
typedef __bf16 bf16;

#define NCOMP  8
#define CIN    512
#define COUT   256
#define HIN    64
#define HOUT   128
#define KTOT   2048            // 4 taps * 512 ci
#define MTILE  128
#define NTILE  128
#define KC     64              // k per stage (2 WMMA k-steps)
#define NCHUNK 32              // KTOT / KC
#define LDSP   72              // padded LDS row in bf16: 144B (TDM pad 16B/128B)

#define AMAT_BYTES ((size_t)4 * COUT * KTOT * 2)        // 4 MiB
#define BBIG_OFF   AMAT_BYTES
#define XH_OFF     (AMAT_BYTES + 4096)
#define XELEMS     ((size_t)8 * CIN * HIN * HIN)        // 16,777,216

// Build signed Cayley-Dickson component table in LDS.
__device__ __forceinline__ void build_comp_lds(int (*sA)[8]) {
    if (threadIdx.x == 0) {
        sA[0][0] = 0;
        for (int m = 1; m < NCOMP; m <<= 1)
            for (int r = 0; r < m; ++r)
                for (int c = 0; c < m; ++c) {
                    int v = sA[r][c];
                    sA[r][c + m]     = -(v + m);
                    sA[r + m][c]     =  v + m;
                    sA[r + m][c + m] =  v;
                }
    }
    __syncthreads();
}

__device__ __forceinline__ int kpos(int parity_bit, int t) {
    return parity_bit ? (t ? 2 : 0) : (t ? 3 : 1);
}

// ---------------------------------------------------------------------------
__global__ __launch_bounds__(256) void prep_xcvt(
    const float* __restrict__ x, bf16* __restrict__ xh) {
    const size_t i = (size_t)blockIdx.x * 256 + threadIdx.x;   // 4 elems each
    const float4 v = ((const float4*)x)[i];
    ((v4bf*)xh)[i] = (v4bf){(bf16)v.x, (bf16)v.y, (bf16)v.z, (bf16)v.w};
}

__global__ __launch_bounds__(256) void prep_weights(
    const float* __restrict__ W, bf16* __restrict__ Amat) {
    __shared__ int sA[8][8];
    build_comp_lds(sA);

    const int e = blockIdx.x * 256 + threadIdx.x;   // 4*256*2048 total
    const int par    = e >> 19;
    const int rem    = e & ((1 << 19) - 1);
    const int co_big = rem >> 11;
    const int k      = rem & 2047;
    const int tap    = k >> 9;
    const int ci_big = k & 511;
    const int py = par >> 1, px = par & 1;
    const int ty = tap >> 1, tx = tap & 1;
    const int kh = kpos(py, ty), kw = kpos(px, tx);

    const int i = co_big >> 5, co_loc = co_big & 31;   // c_out = 32
    const int j = ci_big >> 6, ci_loc = ci_big & 63;   // c_in  = 64
    const int ent = sA[i][j];
    const float sg = (ent < 0) ? -1.0f : 1.0f;
    const int idx = (ent < 0) ? -ent : ent;

    const float w = W[(((size_t)idx * 64 + ci_loc) * 32 + co_loc) * 16 + kh * 4 + kw];
    Amat[e] = (bf16)(sg * w);
}

__global__ __launch_bounds__(256) void prep_bias(
    const float* __restrict__ b, float* __restrict__ bbig) {
    __shared__ int sA[8][8];
    build_comp_lds(sA);
    const int co_big = threadIdx.x;
    const int i = co_big >> 5, co_loc = co_big & 31;
    float acc = 0.f;
    for (int j = 0; j < NCOMP; ++j) {
        int ent = sA[i][j];
        float sg = (ent < 0) ? -1.0f : 1.0f;
        int idx = (ent < 0) ? -ent : ent;
        acc += sg * b[idx * 32 + co_loc];
    }
    bbig[co_big] = acc;
}

// ---------------------------------------------------------------------------
__global__ __launch_bounds__(256) void hconv_wmma(
    const bf16* __restrict__ xh, const bf16* __restrict__ Amat,
    const float* __restrict__ bbig, float* __restrict__ out) {
    __shared__ __align__(16) bf16 As[2][MTILE][LDSP];
    __shared__ __align__(16) bf16 Bs[2][NTILE][LDSP];

    const int nblk = blockIdx.x;            // 0..255 : (image, y-pair)
    const int mblk = blockIdx.y;            // 0..1
    const int par  = blockIdx.z;            // 0..3   : py*2+px
    const int py = par >> 1, px = par & 1;

    const int nimg    = nblk >> 5;
    const int y0      = (nblk & 31) * 2;    // input rows y0, y0+1
    const int co_base = mblk * MTILE;

    const int tid  = threadIdx.x;
    const int lane = tid & 31;
    const int wv   = tid >> 5;
    const int wm   = wv & 1;                // M half (64 rows)
    const int wn   = wv >> 1;               // N quarter (32 pixels)

    const int dyv[2] = { py ? 1 : 0, py ? 0 : -1 };
    const int dxv[2] = { px ? 1 : 0, px ? 0 : -1 };

    v8f acc[4][2];
#pragma unroll
    for (int mf = 0; mf < 4; ++mf)
#pragma unroll
        for (int nf = 0; nf < 2; ++nf)
            acc[mf][nf] = (v8f){0.f, 0.f, 0.f, 0.f, 0.f, 0.f, 0.f, 0.f};

    const bf16* Ap = Amat + (size_t)par * COUT * KTOT;

    // --- TDM: async A tile (128 co x 64 k bf16) into padded LDS rows -------
    auto tdmA = [&](int buf, int tap, int ci0) {
        const int kbase = tap * CIN + ci0;
        const unsigned long long ga =
            (unsigned long long)(uintptr_t)(Ap + (size_t)co_base * KTOT + kbase);
        const unsigned ldsa = (unsigned)(uintptr_t)(&As[buf][0][0]);
        // Group0: count=1 | lds_addr | global_addr | type=2
        u32x4 g0 = { 1u, ldsa, (unsigned)ga,
                     (unsigned)((ga >> 32) & 0x01FFFFFFu) | (2u << 30) };
        // Group1: data_size=2B, pad 4 DW every 32 DW, dims/strides
        //  w0: ds=1<<16 | pad_en 1<<20 | pad_interval 4<<22 | pad_amount 3<<25
        i32x8 g1 = { 0x07110000,
                     0x08000000,          // tensor_dim0 = 2048 (bits 79:48)
                     0x01000000,          // tensor_dim1 = 256  (bits 111:80)
                     0x00400000,          // tile_dim0   = 64   (bits 127:112)
                     128,                 // tile_dim1   = 128  (bits 143:128)
                     2048,                // tensor_dim0_stride (bits 191:160)
                     0, 0 };
        i32x4 z4 = {0, 0, 0, 0};
#if defined(__clang_major__) && (__clang_major__ >= 23)
        i32x8 z8 = {0, 0, 0, 0, 0, 0, 0, 0};
        __builtin_amdgcn_tensor_load_to_lds(g0, g1, z4, z4, z8, 0);
#else
        __builtin_amdgcn_tensor_load_to_lds(g0, g1, z4, z4, 0);
#endif
    };

    // --- B tile: 64 ci x (2 rows x 64 x) bf16, transposed to [pixel][k] ----
    auto stageB = [&](int buf, int tap, int ci0) {
        const int ddy = dyv[tap >> 1];
        const int ddx = dxv[tap & 1];
        const int xp  = tid & 63;
        const int grp = tid >> 6;                 // 0..3
        const int iw  = xp + ddx;
        const bool wok = (unsigned)iw < (unsigned)HIN;
#pragma unroll 8
        for (int it = 0; it < 32; ++it) {
            const int pr  = grp * 32 + it;        // 0..127
            const int cil = pr >> 1;
            const int row = pr & 1;
            const int ih  = y0 + row + ddy;
            bf16 v = (bf16)0.f;
            if (wok && (unsigned)ih < (unsigned)HIN)
                v = xh[(((size_t)nimg * CIN + ci0 + cil) * HIN + ih) * HIN + iw];
            Bs[buf][row * 64 + xp][cil] = v;
        }
    };

    // --- fragments per ISA 16-bit layouts + 16 WMMAs per stage -------------
    auto computeC = [&](int buf) {
        const int khalf = lane >> 4;
        const int l16   = lane & 15;
#pragma unroll
        for (int ks = 0; ks < 2; ++ks) {
            v16bf afr[4], bfr[2];
#pragma unroll
            for (int mf = 0; mf < 4; ++mf) {
                const bf16* p = &As[buf][wm * 64 + mf * 16 + l16][ks * 32 + khalf * 8];
                ((uint4*)&afr[mf])[0] = *(const uint4*)p;
                ((uint4*)&afr[mf])[1] = *(const uint4*)(p + 16);
            }
#pragma unroll
            for (int nf = 0; nf < 2; ++nf) {
                const bf16* p = &Bs[buf][wn * 32 + nf * 16 + l16][ks * 32 + khalf * 8];
                ((uint4*)&bfr[nf])[0] = *(const uint4*)p;
                ((uint4*)&bfr[nf])[1] = *(const uint4*)(p + 16);
            }
#pragma unroll
            for (int mf = 0; mf < 4; ++mf)
#pragma unroll
                for (int nf = 0; nf < 2; ++nf)
                    acc[mf][nf] = __builtin_amdgcn_wmma_f32_16x16x32_bf16(
                        false, afr[mf], false, bfr[nf],
                        (short)0, acc[mf][nf], false, false);
        }
    };

    // --- pipelined main loop ----------------------------------------------
    int buf = 0;
    stageB(0, 0, 0);
    if (wv == 0) {
        tdmA(0, 0, 0);
        __builtin_amdgcn_s_wait_tensorcnt(0);
    }
    __syncthreads();

    for (int c = 0; c < NCHUNK; ++c) {
        const int nc = c + 1;
        if (nc < NCHUNK) {
            stageB(buf ^ 1, nc >> 3, (nc & 7) * KC);
            if (wv == 0) tdmA(buf ^ 1, nc >> 3, (nc & 7) * KC);
        }
        computeC(buf);
        if (wv == 0) __builtin_amdgcn_s_wait_tensorcnt(0);
        __syncthreads();
        buf ^= 1;
    }

    // --- epilogue: C layout -> NCHW output, fused bias ---------------------
    const int khalf = lane >> 4;
    const int l16   = lane & 15;
#pragma unroll
    for (int nf = 0; nf < 2; ++nf) {
        const int nloc = wn * 32 + nf * 16 + l16;
        const int row  = nloc >> 6;
        const int xp   = nloc & 63;
        const int oh   = 2 * (y0 + row) + py;
        const int ow   = 2 * xp + px;
#pragma unroll
        for (int mf = 0; mf < 4; ++mf) {
#pragma unroll
            for (int v = 0; v < 8; ++v) {
                const int co = co_base + wm * 64 + mf * 16 + khalf * 8 + v;
                out[(((size_t)nimg * COUT + co) * HOUT + oh) * HOUT + ow] =
                    acc[mf][nf][v] + bbig[co];
            }
        }
    }
}

// ---------------------------------------------------------------------------
extern "C" void kernel_launch(void* const* d_in, const int* in_sizes, int n_in,
                              void* d_out, int out_size, void* d_ws, size_t ws_size,
                              hipStream_t stream) {
    const float* x = (const float*)d_in[0];
    const float* W = (const float*)d_in[1];
    const float* b = (const float*)d_in[2];
    float* out = (float*)d_out;

    bf16*  Amat = (bf16*)d_ws;
    float* bbig = (float*)((char*)d_ws + BBIG_OFF);
    bf16*  xh   = (bf16*)((char*)d_ws + XH_OFF);

    // 1) x fp32 -> bf16 (vectorized, 4 elems/thread)
    prep_xcvt<<<(unsigned)(XELEMS / 4 / 256), 256, 0, stream>>>(x, xh);
    // 2) pack signed/permuted weights as bf16 GEMM operand per parity class
    prep_weights<<<(4 * COUT * KTOT) / 256, 256, 0, stream>>>(W, Amat);
    // 3) fuse bias through the component table
    prep_bias<<<1, 256, 0, stream>>>(b, bbig);
    // 4) 4 parity-class GEMMs, M=256 x N=32768 x K=2048 each
    hconv_wmma<<<dim3(256, 2, 4), 256, 0, stream>>>(xh, Amat, bbig, out);
    (void)in_sizes; (void)n_in; (void)out_size; (void)ws_size;
}
